// CausalSelfAttention_7937099563056
// MI455X (gfx1250) — matrix operand order repaired and checked by reading, not compiled, not run on hardware
//
#include <hip/hip_runtime.h>
#include <hip/hip_bf16.h>
#include <stdint.h>

typedef __attribute__((ext_vector_type(16))) __bf16 v16bf;
typedef __attribute__((ext_vector_type(8)))  __bf16 v8bf;
typedef __attribute__((ext_vector_type(8)))  float  v8f;

#define B_    4
#define T_    2048
#define H_    16
#define D_    64
#define DM_   1024
#define QKVN_ 3072

// ---- WMMA operand loaders (bf16, wave32) -----------------------------------
// A-matrix 16x32 (MxK), row-major source with 'stride' elements per row.
// ISA layout: lanes 0-15 M=lane hold K={0..7,16..23}; lanes 16-31 M=lane-16
// hold K={8..15,24..31}.  -> two contiguous 16B loads per lane.
__device__ __forceinline__ v16bf load_a16(const __bf16* p, int stride, int lane) {
  const int half = (lane >> 4) & 1;
  const int m    = lane & 15;
  const __bf16* r = p + (size_t)m * stride + 8 * half;
  v8bf lo = *(const v8bf*)(r);        // K = 8*half .. 8*half+7
  v8bf hi = *(const v8bf*)(r + 16);   // K = 16+8*half .. 23+8*half
  v16bf a;
#pragma unroll
  for (int i = 0; i < 8; ++i) { a[i] = lo[i]; a[i + 8] = hi[i]; }
  return a;
}

// B-matrix 32x16 (KxN), row-major source: lane N = lane&15, elements K={8h..8h+7,16+8h..23+8h}.
__device__ __forceinline__ v16bf load_b16(const __bf16* p, int stride, int lane) {
  const int half = (lane >> 4) & 1;
  const int n    = lane & 15;
  v16bf b;
#pragma unroll
  for (int i = 0; i < 8; ++i) {
    b[i]     = p[(size_t)(8 * half + i) * stride + n];
    b[i + 8] = p[(size_t)(16 + 8 * half + i) * stride + n];
  }
  return b;
}

// ---- elementwise f32 -> bf16 ------------------------------------------------
__global__ void f32_to_bf16(const float* __restrict__ in, __bf16* __restrict__ out, int n) {
  int i = blockIdx.x * blockDim.x + threadIdx.x;
  if (i < n) out[i] = (__bf16)in[i];
}

// ---- GEMM: C(MxN,f32) = A(MxK,bf16) * B(KxN,bf16) --------------------------
// Workgroup = 8 waves = 128x64 C tile. B tile (32x64 per K-chunk) staged into
// LDS with async copies (ASYNCcnt), double-buffered, shared by all 8 waves.
__global__ __launch_bounds__(256) void gemm_bf16_wmma(
    const __bf16* __restrict__ A, const __bf16* __restrict__ Bm,
    float* __restrict__ C, int M, int N, int K) {
  __shared__ __bf16 Blds[2][32 * 64];          // 2 x 4KB double buffer
  const int tid  = threadIdx.x;
  const int lane = tid & 31;
  const int wave = tid >> 5;
  const int m0 = blockIdx.y * 128 + wave * 16;
  const int n0 = blockIdx.x * 64;

  // cooperative async stage of B[k0..k0+31][n0..n0+63]: 256 threads x 16B
  const int brow = tid >> 3, bseg = tid & 7;   // row 0..31, 16B segment 0..7
  auto stage = [&](int k0, int buf) {
    const __bf16* gp = Bm + (size_t)(k0 + brow) * N + n0 + bseg * 8;
    unsigned long long ga = (unsigned long long)(uintptr_t)gp;
    unsigned la = (unsigned)(uintptr_t)(&Blds[buf][brow * 64 + bseg * 8]);
    asm volatile("global_load_async_to_lds_b128 %0, %1, off"
                 :: "v"(la), "v"(ga) : "memory");
  };

  v8f acc0 = {}, acc1 = {}, acc2 = {}, acc3 = {};
  const __bf16* arow = A + (size_t)m0 * K;

  stage(0, 0);
  asm volatile("s_wait_asynccnt 0x0" ::: "memory");
  __syncthreads();

  int buf = 0;
  for (int k0 = 0; k0 < K; k0 += 32) {
    if (k0 + 32 < K) stage(k0 + 32, buf ^ 1);  // prefetch next chunk
    v16bf a = load_a16(arow + k0, K, lane);
    const __bf16* bl = &Blds[buf][0];          // ds_load path
    v16bf b0 = load_b16(bl,      64, lane);
    v16bf b1 = load_b16(bl + 16, 64, lane);
    v16bf b2 = load_b16(bl + 32, 64, lane);
    v16bf b3 = load_b16(bl + 48, 64, lane);
    acc0 = __builtin_amdgcn_wmma_f32_16x16x32_bf16(false, a, false, b0, (short)0, acc0, false, false);
    acc1 = __builtin_amdgcn_wmma_f32_16x16x32_bf16(false, a, false, b1, (short)0, acc1, false, false);
    acc2 = __builtin_amdgcn_wmma_f32_16x16x32_bf16(false, a, false, b2, (short)0, acc2, false, false);
    acc3 = __builtin_amdgcn_wmma_f32_16x16x32_bf16(false, a, false, b3, (short)0, acc3, false, false);
    asm volatile("s_wait_asynccnt 0x0" ::: "memory"); // prefetch landed
    __syncthreads();                                  // all waves done reading buf
    buf ^= 1;
  }

  const int half = lane >> 4, n = lane & 15;
#pragma unroll
  for (int r = 0; r < 8; ++r) {
    int row = m0 + r + 8 * half;               // C layout: VGPR r -> M = r + 8*half
    float* c = C + (size_t)row * N + n0 + n;
    c[0] = acc0[r]; c[16] = acc1[r]; c[32] = acc2[r]; c[48] = acc3[r];
  }
}

// ---- RoPE + split qkv(b,t,3,h,64 f32) -> Qb(b,h,t,64) Kt(b,h,64,t) Vb ------
__global__ void rope_split(const float* __restrict__ qkv,
                           __bf16* __restrict__ Qb, __bf16* __restrict__ Kt,
                           __bf16* __restrict__ Vb) {
  int idx = blockIdx.x * blockDim.x + threadIdx.x;   // (((b*H+h)*T+t)*64+d)
  int d = idx & 63;
  int t = (idx >> 6) & (T_ - 1);
  int h = (idx >> 17) & (H_ - 1);
  int b = idx >> 21;
  size_t qo = (((size_t)(b * T_ + t) * 3 + 0) * H_ + h) * D_;
  size_t ko = (((size_t)(b * T_ + t) * 3 + 1) * H_ + h) * D_;
  size_t vo = (((size_t)(b * T_ + t) * 3 + 2) * H_ + h) * D_;
  float q = qkv[qo + d], qp = qkv[qo + (d ^ 32)];
  float k = qkv[ko + d], kp = qkv[ko + (d ^ 32)];
  float v = qkv[vo + d];
  int j = d & 31;
  float ang = (float)t * __powf(10000.0f, -(float)j * (1.0f / 32.0f));
  float sn, cs; __sincosf(ang, &sn, &cs);
  float sg = (d < 32) ? -1.0f : 1.0f;     // rotate_half sign
  float qr = q * cs + sg * qp * sn;
  float kr = k * cs + sg * kp * sn;
  size_t bh = (size_t)(b * H_ + h);
  Qb[(bh * T_ + t) * D_ + d] = (__bf16)qr;
  Kt[(bh * D_ + d) * T_ + t] = (__bf16)kr;     // pre-transposed for QK^T B-operand
  Vb[(bh * T_ + t) * D_ + d] = (__bf16)v;
}

// ---- flash attention: one wave = 16 query rows x full head_dim -------------
__global__ __launch_bounds__(32) void attn_wmma(
    const __bf16* __restrict__ Qb, const __bf16* __restrict__ Ktb,
    const __bf16* __restrict__ Vb, __bf16* __restrict__ Att) {
  __shared__ __bf16 Pl[16 * 32];          // P staging: C-layout f32 -> A-layout bf16
  const int lane = threadIdx.x;
  const int half = lane >> 4, nn = lane & 15;
  const int nqt = T_ / 16;
  int qt = blockIdx.x % nqt;
  int h  = (blockIdx.x / nqt) % H_;
  int b  = blockIdx.x / (nqt * H_);
  const int q0 = qt * 16;
  size_t bh = (size_t)(b * H_ + h);
  const __bf16* Q = Qb  + bh * T_ * D_;
  const __bf16* K = Ktb + bh * D_ * T_;   // (64, T) row-major
  const __bf16* V = Vb  + bh * T_ * D_;

  v16bf qa0 = load_a16(Q + q0 * D_,      D_, lane);   // d 0..31
  v16bf qa1 = load_a16(Q + q0 * D_ + 32, D_, lane);   // d 32..63

  v8f o0 = {}, o1 = {}, o2 = {}, o3 = {};
  float mrow[8], lrow[8];
#pragma unroll
  for (int r = 0; r < 8; ++r) { mrow[r] = -1e30f; lrow[r] = 0.0f; }

  const float scale = 0.125f;             // 1/sqrt(64)
  const int kv_end = q0 + 16;             // causal bound (exclusive)
  for (int kv = 0; kv < kv_end; kv += 32) {
    // S tile 16x32 = Q(16x64) . K^T(64x32), two 16-col halves
    v8f s0 = {}, s1 = {};
    {
      v16bf b0 = load_b16(K + kv,           T_, lane);   // rows=d(0..31), cols kv..kv+15
      v16bf b1 = load_b16(K + 32 * T_ + kv, T_, lane);   // rows=d(32..63)
      s0 = __builtin_amdgcn_wmma_f32_16x16x32_bf16(false, qa0, false, b0, (short)0, s0, false, false);
      s0 = __builtin_amdgcn_wmma_f32_16x16x32_bf16(false, qa1, false, b1, (short)0, s0, false, false);
    }
    if (kv + 16 < kv_end) {               // uniform per-wave branch (EXEC stays full)
      v16bf b0 = load_b16(K + kv + 16,           T_, lane);
      v16bf b1 = load_b16(K + 32 * T_ + kv + 16, T_, lane);
      s1 = __builtin_amdgcn_wmma_f32_16x16x32_bf16(false, qa0, false, b0, (short)0, s1, false, false);
      s1 = __builtin_amdgcn_wmma_f32_16x16x32_bf16(false, qa1, false, b1, (short)0, s1, false, false);
    }
    // mask + scale, new row max (row = q0 + r + 8*half, col = kv + 16*nh + nn)
    float mnew[8], rsum[8];
#pragma unroll
    for (int r = 0; r < 8; ++r) {
      int row = q0 + r + 8 * half;
      float x0 = (kv + nn      <= row) ? s0[r] * scale : -3.0e38f;
      float x1 = (kv + 16 + nn <= row) ? s1[r] * scale : -3.0e38f;
      s0[r] = x0; s1[r] = x1;
      mnew[r] = fmaxf(mrow[r], fmaxf(x0, x1));
    }
#pragma unroll
    for (int off = 1; off < 16; off <<= 1)
#pragma unroll
      for (int r = 0; r < 8; ++r)
        mnew[r] = fmaxf(mnew[r], __shfl_xor(mnew[r], off, 32));
    // rescale O/l, exponentiate, stage P to LDS (bf16, row-major 16x32)
#pragma unroll
    for (int r = 0; r < 8; ++r) {
      float alpha = __expf(mrow[r] - mnew[r]);
      float p0 = __expf(s0[r] - mnew[r]);
      float p1 = __expf(s1[r] - mnew[r]);
      o0[r] *= alpha; o1[r] *= alpha; o2[r] *= alpha; o3[r] *= alpha;
      lrow[r] *= alpha;
      rsum[r] = p0 + p1;
      mrow[r] = mnew[r];
      Pl[(r + 8 * half) * 32 + nn]      = (__bf16)p0;
      Pl[(r + 8 * half) * 32 + 16 + nn] = (__bf16)p1;
    }
#pragma unroll
    for (int off = 1; off < 16; off <<= 1)
#pragma unroll
      for (int r = 0; r < 8; ++r)
        rsum[r] += __shfl_xor(rsum[r], off, 32);
#pragma unroll
    for (int r = 0; r < 8; ++r) lrow[r] += rsum[r];

    asm volatile("s_wait_dscnt 0x0" ::: "memory");   // DS stores visible to our reads
    // reload P in A-layout (16x32 bf16)
    v16bf pa;
#pragma unroll
    for (int i = 0; i < 8; ++i) {
      pa[i]     = Pl[nn * 32 + 8 * half + i];        // lanes: M = lane&15
      pa[i + 8] = Pl[nn * 32 + 16 + 8 * half + i];
    }
    // O(16x64) += P(16x32) . V(32x64)
    const __bf16* vb = V + (size_t)kv * D_;          // rows = kv index (lane)
    v16bf v0 = load_b16(vb,      D_, lane);
    v16bf v1 = load_b16(vb + 16, D_, lane);
    v16bf v2 = load_b16(vb + 32, D_, lane);
    v16bf v3 = load_b16(vb + 48, D_, lane);
    o0 = __builtin_amdgcn_wmma_f32_16x16x32_bf16(false, pa, false, v0, (short)0, o0, false, false);
    o1 = __builtin_amdgcn_wmma_f32_16x16x32_bf16(false, pa, false, v1, (short)0, o1, false, false);
    o2 = __builtin_amdgcn_wmma_f32_16x16x32_bf16(false, pa, false, v2, (short)0, o2, false, false);
    o3 = __builtin_amdgcn_wmma_f32_16x16x32_bf16(false, pa, false, v3, (short)0, o3, false, false);
  }
  // normalize and write as (b, t, h, d) bf16 = the (8192 x 1024) GEMM input
#pragma unroll
  for (int r = 0; r < 8; ++r) {
    int row = q0 + r + 8 * half;
    float inv = 1.0f / lrow[r];
    size_t out = ((size_t)(b * T_ + row) * H_ + h) * D_ + nn;
    Att[out]      = (__bf16)(o0[r] * inv);
    Att[out + 16] = (__bf16)(o1[r] * inv);
    Att[out + 32] = (__bf16)(o2[r] * inv);
    Att[out + 48] = (__bf16)(o3[r] * inv);
  }
}

// ---- driver -----------------------------------------------------------------
extern "C" void kernel_launch(void* const* d_in, const int* in_sizes, int n_in,
                              void* d_out, int out_size, void* d_ws, size_t ws_size,
                              hipStream_t stream) {
  const float* x    = (const float*)d_in[0];   // (4,2048,1024)
  const float* Wqkv = (const float*)d_in[1];   // (1024,3072)
  const float* Wout = (const float*)d_in[2];   // (1024,1024)
  float* out = (float*)d_out;                  // (4,2048,1024) f32

  char* ws = (char*)d_ws;
  size_t off = 0;
  auto carve = [&](size_t bytes) -> void* {
    void* p = ws + off;
    off = (off + bytes + 255) & ~(size_t)255;
    return p;
  };
  const int M = B_ * T_;                              // 8192
  __bf16* Xb    = (__bf16*)carve((size_t)M * DM_ * 2);
  __bf16* Wqkvb = (__bf16*)carve((size_t)DM_ * QKVN_ * 2);
  __bf16* Woutb = (__bf16*)carve((size_t)DM_ * DM_ * 2);
  float*  QKV   = (float*) carve((size_t)M * QKVN_ * 4);
  __bf16* Qb    = (__bf16*)carve((size_t)M * DM_ * 2);
  __bf16* Ktb   = (__bf16*)carve((size_t)M * DM_ * 2);
  __bf16* Vb    = (__bf16*)carve((size_t)M * DM_ * 2);
  __bf16* Attb  = (__bf16*)carve((size_t)M * DM_ * 2);

  int nX = M * DM_, nWq = DM_ * QKVN_, nWo = DM_ * DM_;
  f32_to_bf16<<<(nX  + 255) / 256, 256, 0, stream>>>(x,    Xb,    nX);
  f32_to_bf16<<<(nWq + 255) / 256, 256, 0, stream>>>(Wqkv, Wqkvb, nWq);
  f32_to_bf16<<<(nWo + 255) / 256, 256, 0, stream>>>(Wout, Woutb, nWo);

  gemm_bf16_wmma<<<dim3(QKVN_ / 64, M / 128), 256, 0, stream>>>(Xb, Wqkvb, QKV, M, QKVN_, DM_);

  rope_split<<<(B_ * H_ * T_ * D_) / 256, 256, 0, stream>>>(QKV, Qb, Ktb, Vb);

  attn_wmma<<<B_ * H_ * (T_ / 16), 32, 0, stream>>>(Qb, Ktb, Vb, Attb);

  gemm_bf16_wmma<<<dim3(DM_ / 64, M / 128), 256, 0, stream>>>(Attb, Woutb, out, M, DM_, DM_);
}
